// MambaEncoder_62792421867549
// MI455X (gfx1250) — matrix operand order repaired and checked by reading, not compiled, not run on hardware
//
#include <hip/hip_runtime.h>
#include <hip/hip_bf16.h>
#include <math.h>

// ---------------------------------------------------------------------------
// MambaEncoder for MI455X (gfx1250, wave32).
// GEMMs: fp16 A + fp16 pre-transposed padded weights Bt[n][k], f32 accumulate
// via v_wmma_f32_16x16x32_f16. Tile staging uses CDNA5 async global->LDS
// (global_load_async_to_lds_b128, ASYNCcnt) with LDS double-buffering so the
// async engine overlaps the WMMA pipe. Dense 3-tap conv = one K=768 batched
// GEMM over a per-batch zero-padded fp16 buffer (no masking, no beta).
// Selective scan: 1 lane per (batch, channel), 16 states in VGPRs.
// ---------------------------------------------------------------------------

typedef __attribute__((ext_vector_type(16))) _Float16 v16h;
typedef __attribute__((ext_vector_type(8)))  _Float16 v8h;
typedef __attribute__((ext_vector_type(8)))  float    v8f;

#define BSZ  32
#define TSEQ 512
#define DM   256
#define DI   512
#define NS   16
#define NL   4
#define NG   16
#define NH   4
#define HD   64
#define BT   (BSZ * TSEQ)
#define TP2  (TSEQ + 2)

#define LDS_STRIDE 40  // padded LDS row stride in halves: 80B = 20 banks, 16B aligned

__device__ __forceinline__ float siluf(float x) { return x / (1.f + __expf(-x)); }
__device__ __forceinline__ float softplusf(float x) { return x > 20.f ? x : log1pf(__expf(x)); }

// CDNA5 async global->LDS copy, 16B per lane, tracked by ASYNCcnt.
__device__ __forceinline__ void async_ld_b128(unsigned lds, const void* g) {
  asm volatile("global_load_async_to_lds_b128 %0, %1, off"
               :: "v"(lds), "v"(g) : "memory");
}
__device__ __forceinline__ void wait_async0() {
  asm volatile("s_wait_asynccnt 0x0" ::: "memory");
}
__device__ __forceinline__ unsigned lds_addr(const void* p) {
  return (unsigned)(unsigned long long)p;  // low 32 bits = LDS offset
}

// ---------------------------------------------------------------------------
// WMMA GEMM: C[M,N] = act(A[M,K] @ Bt^T + bias)
//  - A fp16 row-major (lda halves), batch stride aBS (grid.z). A must have at
//    least ceil(M/128)*128 allocated rows (loads are unguarded).
//  - Bt fp16 pre-transposed [n][k], ldb = Kpad (K multiple of 32, Npad mult 64)
//  - C f32 or fp16 (OUTH); stores guarded by m<M, n<N.
//  - Block tile 128x64, 8 waves * (32x32) -> 4 wmma / K-step; LDS ping-pong
//    with async staging overlapping WMMA.
// ---------------------------------------------------------------------------
template<int ACT, bool OUTH>
__global__ __launch_bounds__(256) void k_gemm(
    const _Float16* __restrict__ A, int lda, long aBS,
    const _Float16* __restrict__ Bt, int ldb,
    const float* __restrict__ bias,
    void* __restrict__ Cv, int ldc, long cBS,
    int M, int N, int K)
{
  __shared__ _Float16 As[2][128 * LDS_STRIDE];
  __shared__ _Float16 Bs[2][64 * LDS_STRIDE];   // transposed: Bs[n][k]
  const int tid  = threadIdx.x;
  const int row0 = blockIdx.y * 128;
  const int col0 = blockIdx.x * 64;
  const int z    = blockIdx.z;
  A += (long)z * aBS;
  const int lane = tid & 31, wave = tid >> 5;
  const int wm = (wave >> 1) * 32, wn = (wave & 1) * 32;
  const int hi = lane >> 4, ln16 = lane & 15;
  v8f c00 = {}, c01 = {}, c10 = {}, c11 = {};

  // A loader: row = tid>>1 (0..127), 16 halves starting at (tid&1)*16
  const int lrow = tid >> 1;
  const int colb = (tid & 1) * 16;
  const _Float16* ag = A + (long)(row0 + lrow) * lda + colb;
  // B loader: n = tid>>2 (0..63), 8 halves starting at (tid&3)*8
  const int bn  = tid >> 2;
  const int bkc = (tid & 3) * 8;
  const _Float16* bg = Bt + (long)(col0 + bn) * ldb + bkc;

  auto stage = [&](int kt, int buf) {
    unsigned la = lds_addr(&As[buf][lrow * LDS_STRIDE + colb]);
    async_ld_b128(la,      ag + kt);
    async_ld_b128(la + 16, ag + kt + 8);
    unsigned lb = lds_addr(&Bs[buf][bn * LDS_STRIDE + bkc]);
    async_ld_b128(lb, bg + kt);
  };

  stage(0, 0);
  int cur = 0;
  for (int kt = 0; kt < K; kt += 32) {
    wait_async0();          // our async stores to LDS are visible
    __syncthreads();        // everyone staged cur; everyone done reading cur^1
    if (kt + 32 < K) {
      stage(kt + 32, cur ^ 1);   // overlaps the WMMAs below
      if (kt + 64 < K) {         // deeper prefetch into L2 (global_prefetch_b8)
        __builtin_prefetch(ag + kt + 64, 0, 1);
        __builtin_prefetch(bg + kt + 64, 0, 1);
      }
    }

    // fragments: contiguous ds_load_b128 per ISA VGPR layout
    const _Float16* pa0 = &As[cur][(wm + ln16) * LDS_STRIDE];
    const _Float16* pa1 = pa0 + 16 * LDS_STRIDE;
    v8h a0l = *reinterpret_cast<const v8h*>(pa0 + hi * 8);
    v8h a0h = *reinterpret_cast<const v8h*>(pa0 + 16 + hi * 8);
    v8h a1l = *reinterpret_cast<const v8h*>(pa1 + hi * 8);
    v8h a1h = *reinterpret_cast<const v8h*>(pa1 + 16 + hi * 8);
    const _Float16* pb0 = &Bs[cur][(wn + ln16) * LDS_STRIDE];
    const _Float16* pb1 = pb0 + 16 * LDS_STRIDE;
    v8h b0l = *reinterpret_cast<const v8h*>(pb0 + hi * 8);
    v8h b0h = *reinterpret_cast<const v8h*>(pb0 + 16 + hi * 8);
    v8h b1l = *reinterpret_cast<const v8h*>(pb1 + hi * 8);
    v8h b1h = *reinterpret_cast<const v8h*>(pb1 + 16 + hi * 8);

    v16h a0 = __builtin_shufflevector(a0l, a0h, 0,1,2,3,4,5,6,7,8,9,10,11,12,13,14,15);
    v16h a1 = __builtin_shufflevector(a1l, a1h, 0,1,2,3,4,5,6,7,8,9,10,11,12,13,14,15);
    v16h b0 = __builtin_shufflevector(b0l, b0h, 0,1,2,3,4,5,6,7,8,9,10,11,12,13,14,15);
    v16h b1 = __builtin_shufflevector(b1l, b1h, 0,1,2,3,4,5,6,7,8,9,10,11,12,13,14,15);

    c00 = __builtin_amdgcn_wmma_f32_16x16x32_f16(false, a0, false, b0, (short)0, c00, false, false);
    c01 = __builtin_amdgcn_wmma_f32_16x16x32_f16(false, a0, false, b1, (short)0, c01, false, false);
    c10 = __builtin_amdgcn_wmma_f32_16x16x32_f16(false, a1, false, b0, (short)0, c10, false, false);
    c11 = __builtin_amdgcn_wmma_f32_16x16x32_f16(false, a1, false, b1, (short)0, c11, false, false);
    cur ^= 1;
  }

  // epilogue: lane n = ln16 (+16), VGPR r -> row r + 8*hi (+16)
  const int nA = col0 + wn + ln16;
  const int nB = nA + 16;
  const int mb0 = row0 + wm + hi * 8;
  float biasA = (bias && nA < N) ? bias[nA] : 0.f;
  float biasB = (bias && nB < N) ? bias[nB] : 0.f;
  float*     Cf = OUTH ? nullptr : ((float*)Cv + (long)z * cBS);
  _Float16*  Ch = OUTH ? ((_Float16*)Cv + (long)z * cBS) : nullptr;
#pragma unroll
  for (int r = 0; r < 8; ++r) {
#pragma unroll
    for (int half = 0; half < 2; ++half) {
      int m = mb0 + half * 16 + r;
      if (m >= M) continue;
      if (nA < N) {
        float v = (half ? c10[r] : c00[r]) + biasA;
        if (ACT == 2) v = softplusf(v);
        if (OUTH) Ch[(long)m * ldc + nA] = (_Float16)v;
        else      Cf[(long)m * ldc + nA] = v;
      }
      if (nB < N) {
        float v = (half ? c11[r] : c01[r]) + biasB;
        if (ACT == 2) v = softplusf(v);
        if (OUTH) Ch[(long)m * ldc + nB] = (_Float16)v;
        else      Cf[(long)m * ldc + nB] = v;
      }
    }
  }
}

// ---------------------------------------------------------------------------
// Weight prep: dst[n][koff+k] = (f16) src(k,n), zero beyond K/N. ldK = Kpad.
// ---------------------------------------------------------------------------
__global__ void k_prepw(const float* __restrict__ src, long rs, long cs,
                        int K, int N, _Float16* __restrict__ dst, int ldK, int koff,
                        int Npad, int Kblk)
{
  long i = (long)blockIdx.x * 256 + threadIdx.x;
  if (i >= (long)Npad * Kblk) return;
  int k = (int)(i % Kblk);
  int n = (int)(i / Kblk);
  float v = (k < K && n < N) ? src[(long)k * rs + (long)n * cs] : 0.f;
  dst[(long)n * ldK + koff + k] = (_Float16)v;
}

// x [BT,120] f32 -> XH [BT,128] f16 zero-padded
__global__ void k_cvtx(const float* __restrict__ x, _Float16* __restrict__ XH)
{
  long i = (long)blockIdx.x * 256 + threadIdx.x;
  if (i >= (long)BT * 128) return;
  int j = (int)(i & 127);
  long r = i >> 7;
  XH[i] = (j < 120) ? (_Float16)x[r * 120 + j] : (_Float16)0.f;
}

// XD [BT,48] f32 (cols 0..15 = dt) -> XDh [BT,32] f16 zero-padded
__global__ void k_cvtdt(const float* __restrict__ XD, _Float16* __restrict__ XDh)
{
  long i = (long)blockIdx.x * 256 + threadIdx.x;
  if (i >= (long)BT * 32) return;
  int j = (int)(i & 31);
  long r = i >> 5;
  XDh[i] = (j < 16) ? (_Float16)XD[r * 48 + j] : (_Float16)0.f;
}

// zero n halves
__global__ void k_zeroh(_Float16* __restrict__ p, int n)
{
  int i = blockIdx.x * 256 + threadIdx.x;
  if (i < n) p[i] = (_Float16)0.f;
}

// zero the per-batch pad rows (t=-1, t=T) of the padded GN buffer
__global__ void k_zeroedges(_Float16* __restrict__ G)
{
  int i = blockIdx.x * 256 + threadIdx.x;
  if (i >= BSZ * 2 * DM) return;
  int c = i & (DM - 1);
  int e = (i >> 8) & 1;
  int b = i >> 9;
  long row = (long)b * TP2 + (e ? (TSEQ + 1) : 0);
  G[row * DM + c] = (_Float16)0.f;
}

// ---------------------------------------------------------------------------
// LayerNorm over last dim. One wave per row; HALF selects fp16 output.
// ---------------------------------------------------------------------------
template<bool SILU, bool ADDPOS, bool HALF>
__global__ __launch_bounds__(256) void k_ln(
    const float* __restrict__ X, const float* __restrict__ g,
    const float* __restrict__ b, const float* __restrict__ pos,
    float* __restrict__ Yf, _Float16* __restrict__ Yh)
{
  int wave = threadIdx.x >> 5, lane = threadIdx.x & 31;
  long row = (long)blockIdx.x * 8 + wave;
  const float* xr = X + row * DM;
  float v[8], s = 0.f, s2 = 0.f;
#pragma unroll
  for (int j = 0; j < 8; ++j) { v[j] = xr[lane * 8 + j]; s += v[j]; s2 += v[j] * v[j]; }
#pragma unroll
  for (int o = 16; o > 0; o >>= 1) { s += __shfl_xor(s, o); s2 += __shfl_xor(s2, o); }
  float m   = s / DM;
  float var = s2 / DM - m * m;
  float inv = rsqrtf(var + 1e-5f);
  int t = (int)(row % TSEQ);
#pragma unroll
  for (int j = 0; j < 8; ++j) {
    int d = lane * 8 + j;
    float o = (v[j] - m) * inv * g[d] + b[d];
    if (SILU)   o = siluf(o);
    if (ADDPOS) o += pos[t * DM + d];
    if (HALF) Yh[row * DM + d] = (_Float16)o;
    else      Yf[row * DM + d] = o;
  }
}

// GroupNorm over (T, D/NG) per (b, group); writes fp16 into padded [B,T+2,DM].
__global__ __launch_bounds__(256) void k_gn(
    const float* __restrict__ X, const float* __restrict__ g,
    const float* __restrict__ b, _Float16* __restrict__ Yh)
{
  __shared__ float s1[256], s2[256];
  int bb = blockIdx.x / NG, gg = blockIdx.x % NG;
  int tid = threadIdx.x;
  const int GC = DM / NG;  // 16
  float a = 0.f, a2 = 0.f;
  for (int e = tid; e < TSEQ * GC; e += 256) {
    int t = e >> 4, c = gg * GC + (e & 15);
    float v = X[((long)bb * TSEQ + t) * DM + c];
    a += v; a2 += v * v;
  }
  s1[tid] = a; s2[tid] = a2; __syncthreads();
  for (int s = 128; s > 0; s >>= 1) {
    if (tid < s) { s1[tid] += s1[tid + s]; s2[tid] += s2[tid + s]; }
    __syncthreads();
  }
  float m = s1[0] / (TSEQ * GC), var = s2[0] / (TSEQ * GC) - m * m;
  float inv = rsqrtf(var + 1e-5f);
  for (int e = tid; e < TSEQ * GC; e += 256) {
    int t = e >> 4, c = gg * GC + (e & 15);
    float v = X[((long)bb * TSEQ + t) * DM + c];
    Yh[((long)bb * TP2 + 1 + t) * DM + c] = (_Float16)((v - m) * inv * g[c] + b[c]);
  }
}

// H = (maybe silu(In + bias) + H) * mask
template<bool SILUBIAS>
__global__ void k_resmask(const float* __restrict__ In, const float* __restrict__ bias,
                          float* __restrict__ H, const int* __restrict__ lens)
{
  long i = (long)blockIdx.x * 256 + threadIdx.x;
  if (i >= (long)BT * DM) return;
  int d = (int)(i & (DM - 1));
  long r = i >> 8;
  int t = (int)(r % TSEQ), bb = (int)(r / TSEQ);
  float v = In[i];
  if (SILUBIAS) v = siluf(v + bias[d]);
  float msk = (t < lens[bb]) ? 1.f : 0.f;
  H[i] = (v + H[i]) * msk;
}

// depthwise causal conv (k=4) + bias + SiLU; f32 in, fp16 out
__global__ void k_dwconv(const float* __restrict__ XP, int ldx,
                         const float* __restrict__ w, const float* __restrict__ cb,
                         _Float16* __restrict__ XCh)
{
  long i = (long)blockIdx.x * 256 + threadIdx.x;
  if (i >= (long)BT * DI) return;
  int d = (int)(i & (DI - 1));
  long r = i >> 9;
  int t = (int)(r % TSEQ);
  long rb = r - t;
  float acc = cb[d];
#pragma unroll
  for (int k = 0; k < 4; ++k) {
    int ts = t - 3 + k;
    if (ts >= 0) acc += XP[(rb + ts) * (long)ldx + d] * w[d * 4 + k];
  }
  XCh[r * DI + d] = (_Float16)siluf(acc);
}

// O = Y * silu(Z); fp16 out
__global__ void k_mulsilu(const float* __restrict__ Y, const float* __restrict__ Z, int ldz,
                          _Float16* __restrict__ O)
{
  long i = (long)blockIdx.x * 256 + threadIdx.x;
  if (i >= (long)BT * DI) return;
  int d = (int)(i & (DI - 1));
  long r = i >> 9;
  O[i] = (_Float16)(Y[i] * siluf(Z[r * (long)ldz + d]));
}

// fp16 time-flip copy with independent strides
__global__ void k_fliph(const _Float16* __restrict__ In, int ldi,
                        _Float16* __restrict__ Out, int ldo)
{
  long i = (long)blockIdx.x * 256 + threadIdx.x;
  if (i >= (long)BT * DM) return;
  int d = (int)(i & (DM - 1));
  long r = i >> 8;
  int t = (int)(r % TSEQ);
  long rb = r - t;
  Out[(rb + t) * (long)ldo + d] = In[(rb + (TSEQ - 1 - t)) * (long)ldi + d];
}

// ---------------------------------------------------------------------------
// Selective scan. grid = B, block = 512 (= D_INNER). One lane per channel d;
// 16 h-states in VGPRs; B/C vectors staged through LDS in 64-step chunks.
// ---------------------------------------------------------------------------
__global__ __launch_bounds__(512) void k_scan(
    const _Float16* __restrict__ XCh, const float* __restrict__ DT,
    const float* __restrict__ XDBL, const float* __restrict__ Alog,
    const float* __restrict__ Dsk, float* __restrict__ YO)
{
  __shared__ float bc[64 * 32];
  int bb = blockIdx.x, d = threadIdx.x;
  float h[NS], Ar[NS];
#pragma unroll
  for (int n = 0; n < NS; ++n) { h[n] = 0.f; Ar[n] = -__expf(Alog[d * NS + n]); }
  float Dk = Dsk[d];
  for (int tc = 0; tc < TSEQ; tc += 64) {
#pragma unroll
    for (int q = 0; q < 4; ++q) {
      int e = q * 512 + d;
      int ti = e >> 5, j = e & 31;
      bc[e] = XDBL[((long)bb * TSEQ + tc + ti) * 48 + 16 + j];
    }
    __syncthreads();
    for (int ti = 0; ti < 64; ++ti) {
      long ro = ((long)bb * TSEQ + tc + ti) * DI + d;
      float xt = (float)XCh[ro], dtt = DT[ro];
      float dx = dtt * xt, y = 0.f;
      const float* p = &bc[ti * 32];
#pragma unroll
      for (int n = 0; n < NS; ++n) {
        h[n] = h[n] * __expf(dtt * Ar[n]) + dx * p[n];
        y += h[n] * p[16 + n];
      }
      YO[ro] = y + xt * Dk;
    }
    __syncthreads();
  }
}

// pooled query head: QH = pool_q @ Wq + bq
__global__ void k_qhead(const float* __restrict__ pq, const float* __restrict__ Wq,
                        const float* __restrict__ bq, float* __restrict__ QH)
{
  int n = threadIdx.x;
  float a = bq[n];
  for (int k = 0; k < DM; ++k) a += pq[k] * Wq[k * DM + n];
  QH[n] = a;
}

// attention pooling: one block per (b, head); softmax over T; fp16 out
__global__ __launch_bounds__(512) void k_attnpool(
    const float* __restrict__ QH, const float* __restrict__ Kb,
    const float* __restrict__ Vb, const int* __restrict__ lens,
    _Float16* __restrict__ POh)
{
  __shared__ float sc[TSEQ];
  __shared__ float red[TSEQ];
  int bb = blockIdx.x / NH, hh = blockIdx.x % NH;
  int t = threadIdx.x;
  float s = 0.f;
  const float* kp = Kb + ((long)bb * TSEQ + t) * DM + hh * HD;
  for (int dd = 0; dd < HD; ++dd) s += QH[hh * HD + dd] * kp[dd];
  s *= 0.125f;  // 1/sqrt(64)
  if (t >= lens[bb]) s = -1e9f;
  sc[t] = s; red[t] = s; __syncthreads();
  for (int o = 256; o > 0; o >>= 1) { if (t < o) red[t] = fmaxf(red[t], red[t + o]); __syncthreads(); }
  float mx = red[0]; __syncthreads();
  float e = __expf(s - mx);
  sc[t] = e; red[t] = e; __syncthreads();
  for (int o = 256; o > 0; o >>= 1) { if (t < o) red[t] += red[t + o]; __syncthreads(); }
  float inv = 1.f / red[0]; __syncthreads();
  sc[t] *= inv; __syncthreads();
  if (t < HD) {
    float o = 0.f;
    for (int tt = 0; tt < TSEQ; ++tt)
      o += sc[tt] * Vb[((long)bb * TSEQ + tt) * DM + hh * HD + t];
    POh[bb * DM + hh * HD + t] = (_Float16)o;
  }
}

// ---------------------------------------------------------------------------
// Workspace layout
// ---------------------------------------------------------------------------
constexpr long OF_H   = 0;                               // residual stream f32
constexpr long OF_S1  = OF_H  + (long)BT * DM;           // conv/comb out f32
constexpr long OF_XZ  = OF_S1 + (long)BT * DM;           // in_proj out f32
constexpr long OF_XD  = OF_XZ + (long)BT * 2 * DI;       // x_proj out f32
constexpr long OF_DT  = OF_XD + (long)BT * 48;
constexpr long OF_YS  = OF_DT + (long)BT * DI;
constexpr long OF_KB  = OF_YS + (long)BT * DI;
constexpr long OF_VB  = OF_KB + (long)BT * DM;
constexpr long OF_QH  = OF_VB + (long)BT * DM;
constexpr long F32_TOTAL = OF_QH + 256;

constexpr long OH_XH  = 0;                               // fp16 input, [BT,128]
constexpr long OH_S0  = OH_XH + (long)BT * 128;          // ln out fp16
constexpr long OH_S1  = OH_S0 + (long)BT * DM;           // flipped hn fp16
constexpr long OH_GN  = OH_S1 + (long)BT * DM;           // padded gn out [B,T+2,DM]
constexpr long OH_XC  = OH_GN + (long)BSZ * TP2 * DM;    // dwconv out fp16
constexpr long OH_XD  = OH_XC + (long)BT * DI;           // dt slice fp16 [BT,32]
constexpr long OH_GT  = OH_XD + (long)BT * 32;           // gate out fp16
constexpr long OH_CAT = OH_GT + (long)BT * DI;           // concat fp16 [BT,512]
constexpr long OH_YT  = OH_CAT + (long)BT * 2 * DM;      // bwd out fp16
constexpr long OH_PO  = OH_YT + (long)BT * DM;           // padded to 128 rows
constexpr long OH_PL  = OH_PO + 128L * DM;               // padded to 128 rows
constexpr long OH_WT  = OH_PL + 128L * DM;               // pre-transposed weights

static void gemmh(hipStream_t st, const _Float16* A, int lda, long aBS,
                  const _Float16* Bt, int ldb, const float* bias,
                  void* C, int ldc, long cBS, int M, int N, int K,
                  int batches = 1, int act = 0, bool outh = false)
{
  dim3 g((N + 63) / 64, (M + 127) / 128, batches), b(256);
  if (act == 2)
    k_gemm<2, false><<<g, b, 0, st>>>(A, lda, aBS, Bt, ldb, bias, C, ldc, cBS, M, N, K);
  else if (outh)
    k_gemm<0, true><<<g, b, 0, st>>>(A, lda, aBS, Bt, ldb, bias, C, ldc, cBS, M, N, K);
  else
    k_gemm<0, false><<<g, b, 0, st>>>(A, lda, aBS, Bt, ldb, bias, C, ldc, cBS, M, N, K);
}

static void prepw(hipStream_t st, const float* src, long rs, long cs,
                  int K, int N, _Float16* dst, int ldK, int koff, int Npad, int Kblk)
{
  long tot = (long)Npad * Kblk;
  k_prepw<<<(int)((tot + 255) / 256), 256, 0, st>>>(src, rs, cs, K, N, dst, ldK, koff, Npad, Kblk);
}

// per-direction mamba weights (pre-transposed fp16)
struct MW { _Float16 *inp, *xp, *dt, *outp; };

static void run_mamba(hipStream_t st, const _Float16* hnh, void* const* d_in, int pb,
                      float* W, _Float16* HB, const MW& mw, _Float16* outh, int ldo)
{
  auto P = [&](int i) { return (const float*)d_in[i]; };
  float* XZ  = W + OF_XZ;
  float* XD  = W + OF_XD;
  float* DTb = W + OF_DT;
  float* YS  = W + OF_YS;
  _Float16* XCh = HB + OH_XC;
  _Float16* XDh = HB + OH_XD;
  _Float16* GTh = HB + OH_GT;
  const int NE = (BT * DI + 255) / 256;
  gemmh(st, hnh, DM, 0, mw.inp, DM, nullptr, XZ, 2 * DI, 0, BT, 2 * DI, DM);      // in_proj
  k_dwconv<<<NE, 256, 0, st>>>(XZ, 2 * DI, P(pb + 1), P(pb + 2), XCh);            // causal dwconv+silu
  gemmh(st, XCh, DI, 0, mw.xp, DI, nullptr, XD, 48, 0, BT, 48, DI);               // x_proj
  k_cvtdt<<<(BT * 32 + 255) / 256, 256, 0, st>>>(XD, XDh);                        // dt slice -> fp16 pad32
  gemmh(st, XDh, 32, 0, mw.dt, 32, P(pb + 5), DTb, DI, 0, BT, DI, 32, 1, 2);      // dt proj + softplus
  k_scan<<<BSZ, 512, 0, st>>>(XCh, DTb, XD, P(pb + 6), P(pb + 7), YS);            // selective scan
  k_mulsilu<<<NE, 256, 0, st>>>(YS, XZ + DI, 2 * DI, GTh);                        // gate
  gemmh(st, GTh, DI, 0, mw.outp, DI, nullptr, outh, ldo, 0, BT, DM, DI, 1, 0, true); // out_proj
}

extern "C" void kernel_launch(void* const* d_in, const int* in_sizes, int n_in,
                              void* d_out, int out_size, void* d_ws, size_t ws_size,
                              hipStream_t stream)
{
  (void)in_sizes; (void)n_in; (void)out_size; (void)ws_size;
  const float* x    = (const float*)d_in[0];
  const int*   lens = (const int*)d_in[1];
  auto P = [&](int i) { return (const float*)d_in[i]; };
  float* W = (float*)d_ws;
  _Float16* HB = (_Float16*)(W + F32_TOTAL);
  float* H   = W + OF_H;
  float* S1f = W + OF_S1;
  float* KB  = W + OF_KB;
  float* VB  = W + OF_VB;
  float* QH  = W + OF_QH;
  _Float16* XHh  = HB + OH_XH;
  _Float16* S0h  = HB + OH_S0;
  _Float16* S1h  = HB + OH_S1;
  _Float16* GNh  = HB + OH_GN;
  _Float16* CATh = HB + OH_CAT;
  _Float16* YTh  = HB + OH_YT;
  _Float16* POh  = HB + OH_PO;
  _Float16* PLh  = HB + OH_PL;
  float* out = (float*)d_out;
  const int ND = (BT * DM + 255) / 256;

  // ---- one-time weight prep: fp16 Bt[n][k], padded ----
  _Float16* WT = HB + OH_WT;
  long wo = 0;
  auto walloc = [&](long n) { _Float16* p = WT + wo; wo += n; return p; };
  _Float16* w_inw = walloc(256L * 128);
  _Float16* w_conv[NL];
  MW mw[NL][2];
  _Float16* w_comb[NL];
  for (int l = 0; l < NL; ++l) {
    w_conv[l] = walloc(256L * 768);
    for (int ddir = 0; ddir < 2; ++ddir) {
      mw[l][ddir].inp  = walloc(1024L * 256);
      mw[l][ddir].xp   = walloc(64L * 512);
      mw[l][ddir].dt   = walloc(512L * 32);
      mw[l][ddir].outp = walloc(256L * 512);
    }
    w_comb[l] = walloc(256L * 512);
  }
  _Float16* w_wk = walloc(256L * 256);
  _Float16* w_wv = walloc(256L * 256);
  _Float16* w_wo = walloc(256L * 256);
  _Float16* w_mu = walloc(256L * 256);
  _Float16* w_lv = walloc(256L * 256);

  prepw(stream, P(2), DM, 1, 120, DM, w_inw, 128, 0, DM, 128);               // in_w [120,256]
  for (int l = 0; l < NL; ++l) {
    int pb = 7 + l * 26;
    for (int kk = 0; kk < 3; ++kk)                                           // conv_w [256,256,3]
      prepw(stream, P(pb + 2) + kk, 3, 3 * DM, DM, DM, w_conv[l], 768, kk * DM, DM, DM);
    for (int ddir = 0; ddir < 2; ++ddir) {
      int mb = pb + 6 + ddir * 9;
      prepw(stream, P(mb + 0), 2 * DI, 1, DM, 2 * DI, mw[l][ddir].inp, DM, 0, 2 * DI, DM);
      prepw(stream, P(mb + 3), 48, 1, DI, 48, mw[l][ddir].xp, DI, 0, 64, DI);
      prepw(stream, P(mb + 4), DI, 1, 16, DI, mw[l][ddir].dt, 32, 0, DI, 32);
      prepw(stream, P(mb + 8), DM, 1, DI, DM, mw[l][ddir].outp, DI, 0, DM, DI);
    }
    prepw(stream, P(pb + 24), DM, 1, 2 * DI, DM, w_comb[l], DI, 0, DM, DI);
  }
  prepw(stream, P(116), DM, 1, DM, DM, w_wk, DM, 0, DM, DM);
  prepw(stream, P(118), DM, 1, DM, DM, w_wv, DM, 0, DM, DM);
  prepw(stream, P(120), DM, 1, DM, DM, w_wo, DM, 0, DM, DM);
  prepw(stream, P(122), DM, 1, DM, DM, w_mu, DM, 0, DM, DM);
  prepw(stream, P(124), DM, 1, DM, DM, w_lv, DM, 0, DM, DM);

  // ---- input embed: (x @ in_w + b) -> LN -> SiLU -> +pos ----
  k_cvtx<<<(BT * 128 + 255) / 256, 256, 0, stream>>>(x, XHh);
  gemmh(stream, XHh, 128, 0, w_inw, 128, P(3), S1f, DM, 0, BT, DM, 128);
  k_ln<true, true, false><<<BT / 8, 256, 0, stream>>>(S1f, P(4), P(5), P(6), H, nullptr);

  for (int l = 0; l < NL; ++l) {
    int pb = 7 + l * 26;
    // Local module: GN -> conv('same') as one batched K=768 GEMM -> silu+res+mask
    k_zeroedges<<<(BSZ * 2 * DM + 255) / 256, 256, 0, stream>>>(GNh);
    k_gn<<<BSZ * NG, 256, 0, stream>>>(H, P(pb + 0), P(pb + 1), GNh);
    gemmh(stream, GNh, DM, (long)TP2 * DM, w_conv[l], 768, nullptr,
          S1f, DM, (long)TSEQ * DM, TSEQ, DM, 768, BSZ);
    k_resmask<true><<<ND, 256, 0, stream>>>(S1f, P(pb + 3), H, lens);
    // Bidirectional Mamba
    k_ln<false, false, true><<<BT / 8, 256, 0, stream>>>(H, P(pb + 4), P(pb + 5), nullptr, nullptr, S0h);
    run_mamba(stream, S0h, d_in, pb + 6, W, HB, mw[l][0], CATh, 2 * DM);     // fwd -> CAT[:,0:256]
    k_fliph<<<ND, 256, 0, stream>>>(S0h, DM, S1h, DM);
    run_mamba(stream, S1h, d_in, pb + 15, W, HB, mw[l][1], YTh, DM);         // bwd on flipped
    k_fliph<<<ND, 256, 0, stream>>>(YTh, DM, CATh + DM, 2 * DM);             // unflip -> CAT[:,256:512]
    gemmh(stream, CATh, 2 * DM, 0, w_comb[l], DI, P(pb + 25), S1f, DM, 0, BT, DM, 2 * DM);
    k_resmask<false><<<ND, 256, 0, stream>>>(S1f, nullptr, H, lens);
  }

  // ---- final LN, attention pooling, heads ----
  k_ln<false, false, true><<<BT / 8, 256, 0, stream>>>(H, P(111), P(112), nullptr, nullptr, S0h);
  k_qhead<<<1, 256, 0, stream>>>(P(113), P(114), P(115), QH);
  gemmh(stream, S0h, DM, 0, w_wk, DM, P(117), KB, DM, 0, BT, DM, DM);
  gemmh(stream, S0h, DM, 0, w_wv, DM, P(119), VB, DM, 0, BT, DM, DM);
  k_zeroh<<<(128 * DM + 255) / 256, 256, 0, stream>>>(POh, 128 * DM);
  k_zeroh<<<(128 * DM + 255) / 256, 256, 0, stream>>>(PLh, 128 * DM);
  k_attnpool<<<BSZ * NH, 512, 0, stream>>>(QH, KB, VB, lens, POh);
  gemmh(stream, POh, DM, 0, w_wo, DM, P(121), PLh, DM, 0, BSZ, DM, DM, 1, 0, true);
  gemmh(stream, PLh, DM, 0, w_mu, DM, P(123), out, DM, 0, BSZ, DM, DM);            // mu
  gemmh(stream, PLh, DM, 0, w_lv, DM, P(125), out + (long)BSZ * DM, DM, 0, BSZ, DM, DM); // logvar
}
